// GSS_30013231464846
// MI455X (gfx1250) — compile-verified, matrix operand order checked by reading, not run on hardware
//
#include <hip/hip_runtime.h>
#include <hip/hip_bf16.h>

typedef __attribute__((ext_vector_type(16))) __bf16 v16bf;
typedef __attribute__((ext_vector_type(8)))  __bf16 v8bf;
typedef __attribute__((ext_vector_type(8)))  float  v8f;
typedef __attribute__((ext_vector_type(4)))  unsigned int u32x4;
typedef __attribute__((ext_vector_type(8)))  int i32x8;
typedef __attribute__((ext_vector_type(4)))  int i32x4;

#if defined(__has_builtin)
#  if __has_builtin(__builtin_amdgcn_tensor_load_to_lds) && __has_builtin(__builtin_amdgcn_s_wait_tensorcnt)
#    define HAS_TDM 1
#  endif
#endif
#ifndef HAS_TDM
#  define HAS_TDM 0
#endif

// Problem dims
constexpr int Bb   = 2;
constexpr int Ll   = 4096;
constexpr int BL   = Bb * Ll;     // 8192
constexpr int DIM  = 1024;
constexpr int HID  = 4096;
constexpr int DSSH = 512;
constexpr int DSSN = 512;
constexpr int N2   = 2 * Ll;      // 8192 FFT length
constexpr int LOG2N2 = 13;

// ---------------------------------------------------------------------------
// LayerNorm over last dim W
// ---------------------------------------------------------------------------
__global__ __launch_bounds__(256) void ln_kernel(
    const float* __restrict__ in, const float* __restrict__ w,
    const float* __restrict__ b, float* __restrict__ out_f32,
    __bf16* __restrict__ out_bf16, int W)
{
    const int row = blockIdx.x;
    const float* x = in + (size_t)row * W;
    const int tid = threadIdx.x;
    float s = 0.f, s2 = 0.f;
    for (int i = tid; i < W; i += 256) { float v = x[i]; s += v; s2 += v * v; }
    for (int off = 16; off; off >>= 1) {
        s  += __shfl_xor(s,  off);
        s2 += __shfl_xor(s2, off);
    }
    __shared__ float ps[8], ps2[8];
    const int wid = tid >> 5, lane = tid & 31;
    if (lane == 0) { ps[wid] = s; ps2[wid] = s2; }
    __syncthreads();
    float st = 0.f, st2 = 0.f;
    #pragma unroll
    for (int i = 0; i < 8; i++) { st += ps[i]; st2 += ps2[i]; }
    const float mean = st / (float)W;
    const float var  = st2 / (float)W - mean * mean;
    const float inv  = rsqrtf(var + 1e-5f);
    for (int i = tid; i < W; i += 256) {
        float v = (x[i] - mean) * inv * w[i] + b[i];
        if (out_f32)  out_f32[(size_t)row * W + i]  = v;
        if (out_bf16) out_bf16[(size_t)row * W + i] = (__bf16)v;
    }
}

// ---------------------------------------------------------------------------
// Tiled transpose: in [R,C] f32 -> out [C,R] (f32 or bf16)
// ---------------------------------------------------------------------------
template <typename TO>
__global__ __launch_bounds__(256) void transpose_kernel(
    const float* __restrict__ in, TO* __restrict__ out, int R, int C)
{
    __shared__ float tile[32][33];
    const int c0 = blockIdx.x * 32, r0 = blockIdx.y * 32;
    const int tx = threadIdx.x;
    for (int j = threadIdx.y; j < 32; j += 8)
        tile[j][tx] = in[(size_t)(r0 + j) * C + c0 + tx];
    __syncthreads();
    for (int j = threadIdx.y; j < 32; j += 8)
        out[(size_t)(c0 + j) * R + r0 + tx] = (TO)tile[tx][j];
}

// ---------------------------------------------------------------------------
// Build CCT [DSSH, 2*DSSN] bf16 : row h = [ CR[h,:] , -CI[h,:] ]
// ---------------------------------------------------------------------------
__global__ __launch_bounds__(256) void cprep_kernel(
    const float* __restrict__ CR, const float* __restrict__ CI,
    __bf16* __restrict__ CCT)
{
    const int i = blockIdx.x * 256 + threadIdx.x;  // 512*1024 total
    const int h = i >> 10, k = i & 1023;
    float v = (k < DSSN) ? CR[h * DSSN + k] : -CI[h * DSSN + (k - DSSN)];
    CCT[i] = (__bf16)v;
}

// ---------------------------------------------------------------------------
// DSS setup: Lam = -exp(LR) + i exp(LI); w = (exp(Lam)-1)/Lam
// ---------------------------------------------------------------------------
__global__ void dss_setup(const float* __restrict__ LR, const float* __restrict__ LI,
                          float* lamR, float* lamI, float* wR, float* wI)
{
    const int n = blockIdx.x * blockDim.x + threadIdx.x;
    if (n >= DSSN) return;
    const float ar = -expf(LR[n]);
    const float ai =  expf(LI[n]);
    const float ex = expf(ar);
    float sa, ca; __sincosf(ai, &sa, &ca);
    const float nr = ex * ca - 1.0f;
    const float ni = ex * sa;
    const float den = ar * ar + ai * ai;
    lamR[n] = ar; lamI[n] = ai;
    wR[n] = (nr * ar + ni * ai) / den;
    wI[n] = (ni * ar - nr * ai) / den;
}

// ---------------------------------------------------------------------------
// Build A_s [Ll, 2*DSSN] bf16 : A_s[l] = [ Re(S*w) , Im(S*w) ],  S = exp(Lam*l)
// ---------------------------------------------------------------------------
__global__ __launch_bounds__(256) void sgen_kernel(
    const float* __restrict__ lamR, const float* __restrict__ lamI,
    const float* __restrict__ wR, const float* __restrict__ wI,
    __bf16* __restrict__ As)
{
    const int l = blockIdx.x;
    const float fl = (float)l;
    for (int n = threadIdx.x; n < DSSN; n += 256) {
        const float e = __expf(lamR[n] * fl);
        float sa, ca; __sincosf(lamI[n] * fl, &sa, &ca);
        const float sr = e * ca, si = e * sa;
        const float tr = sr * wR[n] - si * wI[n];
        const float ti = sr * wI[n] + si * wR[n];
        As[(size_t)l * 1024 + n]        = (__bf16)tr;
        As[(size_t)l * 1024 + DSSN + n] = (__bf16)ti;
    }
}

// ---------------------------------------------------------------------------
// In-LDS radix-2 DIT FFT, N=8192, input pre-bit-reversed, output natural.
// ---------------------------------------------------------------------------
__device__ inline void fft_stages(float2* buf, int tid, float sign)
{
    #pragma unroll 1
    for (int s = 1; s <= LOG2N2; s++) {
        const int m = 1 << s, half = m >> 1;
        const float ang0 = sign * 6.2831853071795864f / (float)m;
        for (int j = tid; j < (N2 >> 1); j += 256) {
            const int pos = j & (half - 1);
            const int blk = j >> (s - 1);
            const int i0 = blk * m + pos, i1 = i0 + half;
            float saa, caa; __sincosf(ang0 * (float)pos, &saa, &caa);
            const float2 u = buf[i0], v = buf[i1];
            const float tr = v.x * caa - v.y * saa;
            const float ti = v.x * saa + v.y * caa;
            buf[i0] = make_float2(u.x + tr, u.y + ti);
            buf[i1] = make_float2(u.x - tr, u.y - ti);
        }
        __syncthreads();
    }
}

__device__ inline int brev13(int i) { return (int)(__brev((unsigned)i) >> (32 - LOG2N2)); }

__global__ __launch_bounds__(256) void fft_k_kernel(
    const float* __restrict__ KT, float2* __restrict__ Kf)
{
    __shared__ float2 buf[N2];
    const int h = blockIdx.x, tid = threadIdx.x;
    for (int i = tid; i < N2; i += 256) {
        const float v = (i < Ll) ? KT[(size_t)h * Ll + i] : 0.f;
        buf[brev13(i)] = make_float2(v, 0.f);
    }
    __syncthreads();
    fft_stages(buf, tid, -1.f);
    for (int i = tid; i < N2; i += 256) Kf[(size_t)h * N2 + i] = buf[i];
}

__global__ __launch_bounds__(256) void fft_conv_kernel(
    const float* __restrict__ xnT, const float2* __restrict__ Kf,
    const float* __restrict__ Dp, float* __restrict__ dT)
{
    __shared__ float2 buf[N2];
    const int bh = blockIdx.x;                 // 0..Bb*DSSH-1
    const int h = bh & (DSSH - 1), bb = bh >> 9;
    const float* xr = xnT + (size_t)h * BL + (size_t)bb * Ll;
    const int tid = threadIdx.x;
    float xv[16];
    for (int i = tid, q = 0; i < N2; i += 256, q++) {
        float v = 0.f;
        if (i < Ll) { v = xr[i]; xv[q] = v; }
        buf[brev13(i)] = make_float2(v, 0.f);
    }
    __syncthreads();
    fft_stages(buf, tid, -1.f);
    for (int i = tid; i < N2; i += 256) {
        const float2 a = buf[i], k = Kf[(size_t)h * N2 + i];
        buf[i] = make_float2(a.x * k.x - a.y * k.y, a.x * k.y + a.y * k.x);
    }
    __syncthreads();
    for (int i = tid; i < N2; i += 256) {
        const int r = brev13(i);
        if (i < r) { float2 t = buf[i]; buf[i] = buf[r]; buf[r] = t; }
    }
    __syncthreads();
    fft_stages(buf, tid, +1.f);
    const float scale = 1.f / (float)N2;
    const float Dh = Dp[h];
    for (int i = tid, q = 0; i < Ll; i += 256, q++)
        dT[(size_t)h * BL + (size_t)bb * Ll + i] = buf[i].x * scale + Dh * xv[q];
}

// ---------------------------------------------------------------------------
// Shared WMMA compute / epilogue helpers for the 128x128 GEMM tile
// ---------------------------------------------------------------------------
__device__ inline void gemm_frag_step(
    const __bf16* __restrict__ AsBuf, const __bf16* __restrict__ BsBuf,
    int aRow0, int bCol0, int fm, int kh, v8f acc[2][4])
{
    v16bf af[2], bfr[4];
    #pragma unroll
    for (int i = 0; i < 2; i++) {
        const int row = aRow0 + i * 16 + fm;
        const v8bf lo = *(const v8bf*)&AsBuf[row * 32 + kh * 8];
        const v8bf hi = *(const v8bf*)&AsBuf[row * 32 + 16 + kh * 8];
        af[i] = __builtin_shufflevector(lo, hi, 0,1,2,3,4,5,6,7,8,9,10,11,12,13,14,15);
    }
    #pragma unroll
    for (int j = 0; j < 4; j++) {
        const int row = bCol0 + j * 16 + fm;
        const v8bf lo = *(const v8bf*)&BsBuf[row * 32 + kh * 8];
        const v8bf hi = *(const v8bf*)&BsBuf[row * 32 + 16 + kh * 8];
        bfr[j] = __builtin_shufflevector(lo, hi, 0,1,2,3,4,5,6,7,8,9,10,11,12,13,14,15);
    }
    #pragma unroll
    for (int i = 0; i < 2; i++)
        #pragma unroll
        for (int j = 0; j < 4; j++)
            acc[i][j] = __builtin_amdgcn_wmma_f32_16x16x32_bf16(
                false, af[i], false, bfr[j], (short)0, acc[i][j], false, false);
}

template <int EPI>
__device__ inline void gemm_epilogue(
    v8f acc[2][4], void* __restrict__ Cout, const __bf16* __restrict__ aux,
    const float* __restrict__ bias, const float* __restrict__ res,
    int m0, int n0, int aRow0, int bCol0, int fm, int kh, int M, int N)
{
    #pragma unroll
    for (int i = 0; i < 2; i++) {
        #pragma unroll
        for (int j = 0; j < 4; j++) {
            const int gr0 = m0 + aRow0 + i * 16 + kh * 8;
            const int gc  = n0 + bCol0 + j * 16 + fm;
            #pragma unroll
            for (int r = 0; r < 8; r++) {
                const int gr = gr0 + r;
                const float v = acc[i][j][r];
                const size_t off = (size_t)gr * N + gc;
                if constexpr (EPI == 0) {
                    ((float*)Cout)[off] = v;
                } else if constexpr (EPI == 1) {
                    ((__bf16*)Cout)[off] = (__bf16)v;
                } else if constexpr (EPI == 2) {
                    ((float*)Cout)[(size_t)gc * M + gr] = v;
                } else if constexpr (EPI == 3) {
                    ((__bf16*)Cout)[off] = (__bf16)(v * (float)aux[off]);
                } else {
                    ((float*)Cout)[off] = v + bias[gc] + res[off];
                }
            }
        }
    }
}

// ---------------------------------------------------------------------------
// WMMA bf16 GEMM, vmem-staged double buffer
// ---------------------------------------------------------------------------
template <int EPI>
__global__ __launch_bounds__(256) void wmma_gemm(
    const __bf16* __restrict__ A, const __bf16* __restrict__ BT,
    void* __restrict__ Cout, const __bf16* __restrict__ aux,
    const float* __restrict__ bias, const float* __restrict__ res,
    int M, int N, int K)
{
    __shared__ __bf16 As[2][128 * 32];
    __shared__ __bf16 Bs[2][128 * 32];
    const int tid = threadIdx.x;
    const int m0 = blockIdx.y * 128;
    const int n0 = blockIdx.x * 128;
    const int lr = tid >> 1;
    const int lc = (tid & 1) * 16;
    const __bf16* Aptr = A  + (size_t)(m0 + lr) * K + lc;
    const __bf16* Bptr = BT + (size_t)(n0 + lr) * K + lc;
    const int nk = K >> 5;

    {
        const uint4 ra = *(const uint4*)Aptr;
        const uint4 rb = *(const uint4*)Bptr;
        *(uint4*)&As[0][lr * 32 + lc] = ra;
        *(uint4*)&Bs[0][lr * 32 + lc] = rb;
    }
    __syncthreads();

    const int wave = tid >> 5, lane = tid & 31;
    const int wm = wave & 3, wn = wave >> 2;
    const int fm = lane & 15, kh = lane >> 4;
    const int aRow0 = wm * 32;
    const int bCol0 = wn * 64;

    const v8f vzero = {};
    v8f acc[2][4];
    #pragma unroll
    for (int i = 0; i < 2; i++)
        #pragma unroll
        for (int j = 0; j < 4; j++) acc[i][j] = vzero;

    for (int ks = 0; ks < nk; ks++) {
        const int buf = ks & 1;
        uint4 ra, rb;
        const bool has = (ks + 1 < nk);
        if (has) {
            ra = *(const uint4*)(Aptr + (size_t)(ks + 1) * 32);
            rb = *(const uint4*)(Bptr + (size_t)(ks + 1) * 32);
        }
        gemm_frag_step(&As[buf][0], &Bs[buf][0], aRow0, bCol0, fm, kh, acc);
        if (has) {
            *(uint4*)&As[buf ^ 1][lr * 32 + lc] = ra;
            *(uint4*)&Bs[buf ^ 1][lr * 32 + lc] = rb;
        }
        __syncthreads();
    }
    gemm_epilogue<EPI>(acc, Cout, aux, bias, res, m0, n0, aRow0, bCol0, fm, kh, M, N);
}

#if HAS_TDM
// ---------------------------------------------------------------------------
// TDM descriptor: 2-D tile (tile_dim0=kElems, tile_dim1=rows) of bf16 from a
// row-major matrix with row stride strideElems, compact into LDS at lds_off.
// D# fields per CDNA5 ISA 8.3/8.4: count=1, type=2, data_size=1 (2 bytes).
// Toolchain uses the 6-arg builtin: (g0, g1, g2, g3, g4, cpol).
// ---------------------------------------------------------------------------
__device__ inline void tdm_load_tile_bf16(unsigned lds_off, const void* gptr,
                                          int rows, int kElems, int strideElems)
{
    const unsigned long long ga = (unsigned long long)(size_t)gptr;
    u32x4 g0;
    g0[0] = 1u;                                              // count=1, is_restore=0
    g0[1] = lds_off;                                         // lds_addr
    g0[2] = (unsigned)(ga & 0xFFFFFFFFu);                    // global_addr lo
    g0[3] = (unsigned)((ga >> 32) & 0x01FFFFFFu) | (2u << 30); // global_addr hi | type=2
    i32x8 g1;
    g1[0] = (1 << 16);                                       // mask=0, data_size=1 (2B)
    g1[1] = (kElems & 0xFFFF) << 16;                         // tensor_dim0 lo16
    g1[2] = ((kElems >> 16) & 0xFFFF) | ((rows & 0xFFFF) << 16);   // dim0 hi | dim1 lo
    g1[3] = ((rows >> 16) & 0xFFFF) | ((kElems & 0xFFFF) << 16);   // dim1 hi | tile_dim0
    g1[4] = (rows & 0xFFFF);                                 // tile_dim1 | tile_dim2=0
    g1[5] = strideElems;                                     // tensor_dim0_stride lo32
    g1[6] = 0;                                               // dim0_stride hi | dim1_stride lo
    g1[7] = 0;
    const i32x4 z4 = {0, 0, 0, 0};
    const i32x8 z8 = {0, 0, 0, 0, 0, 0, 0, 0};
    __builtin_amdgcn_tensor_load_to_lds(g0, g1, z4, z4, z8, 0);
}

// ---------------------------------------------------------------------------
// WMMA bf16 GEMM, TDM-staged double buffer: wave 0 drives the Tensor Data
// Mover for both 128x32 tiles; DMA of the next buffer overlaps WMMA compute.
// ---------------------------------------------------------------------------
template <int EPI>
__global__ __launch_bounds__(256) void wmma_gemm_tdm(
    const __bf16* __restrict__ A, const __bf16* __restrict__ BT,
    void* __restrict__ Cout, const __bf16* __restrict__ aux,
    const float* __restrict__ bias, const float* __restrict__ res,
    int M, int N, int K)
{
    __shared__ __bf16 As[2][128 * 32];
    __shared__ __bf16 Bs[2][128 * 32];
    const int tid = threadIdx.x;
    const int m0 = blockIdx.y * 128;
    const int n0 = blockIdx.x * 128;
    const int nk = K >> 5;

    const int wave = tid >> 5, lane = tid & 31;
    const int wm = wave & 3, wn = wave >> 2;
    const int fm = lane & 15, kh = lane >> 4;
    const int aRow0 = wm * 32;
    const int bCol0 = wn * 64;

    const unsigned ldsA0 = (unsigned)(size_t)&As[0][0];
    const unsigned ldsA1 = (unsigned)(size_t)&As[1][0];
    const unsigned ldsB0 = (unsigned)(size_t)&Bs[0][0];
    const unsigned ldsB1 = (unsigned)(size_t)&Bs[1][0];

    if (wave == 0) {
        tdm_load_tile_bf16(ldsA0, A  + (size_t)m0 * K, 128, 32, K);
        tdm_load_tile_bf16(ldsB0, BT + (size_t)n0 * K, 128, 32, K);
        __builtin_amdgcn_s_wait_tensorcnt(0);
    }
    __syncthreads();

    const v8f vzero = {};
    v8f acc[2][4];
    #pragma unroll
    for (int i = 0; i < 2; i++)
        #pragma unroll
        for (int j = 0; j < 4; j++) acc[i][j] = vzero;

    for (int ks = 0; ks < nk; ks++) {
        const int buf = ks & 1;
        const bool has = (ks + 1 < nk);
        if (has && wave == 0) {
            tdm_load_tile_bf16(buf ? ldsA0 : ldsA1,
                               A + (size_t)m0 * K + (size_t)(ks + 1) * 32, 128, 32, K);
            tdm_load_tile_bf16(buf ? ldsB0 : ldsB1,
                               BT + (size_t)n0 * K + (size_t)(ks + 1) * 32, 128, 32, K);
        }
        gemm_frag_step(&As[buf][0], &Bs[buf][0], aRow0, bCol0, fm, kh, acc);
        if (wave == 0) __builtin_amdgcn_s_wait_tensorcnt(0);
        __syncthreads();
    }
    gemm_epilogue<EPI>(acc, Cout, aux, bias, res, m0, n0, aRow0, bCol0, fm, kh, M, N);
}
#endif  // HAS_TDM

// ---------------------------------------------------------------------------
// Launcher
// ---------------------------------------------------------------------------
extern "C" void kernel_launch(void* const* d_in, const int* in_sizes, int n_in,
                              void* d_out, int out_size, void* d_ws, size_t ws_size,
                              hipStream_t stream)
{
    (void)in_sizes; (void)n_in; (void)out_size; (void)ws_size;
    const float* idx   = (const float*)d_in[0];
    const float* n1_w  = (const float*)d_in[1];
    const float* n1_b  = (const float*)d_in[2];
    const float* Wxg   = (const float*)d_in[3];
    const float* Wdin  = (const float*)d_in[4];
    const float* n2_w  = (const float*)d_in[5];
    const float* n2_b  = (const float*)d_in[6];
    const float* LR    = (const float*)d_in[7];
    const float* LI    = (const float*)d_in[8];
    const float* CR    = (const float*)d_in[9];
    const float* CI    = (const float*)d_in[10];
    const float* Dp    = (const float*)d_in[11];
    const float* Wgate = (const float*)d_in[12];
    const float* Wout  = (const float*)d_in[13];
    const float* bout  = (const float*)d_in[14];
    float* out = (float*)d_out;

    char* wsp = (char*)d_ws;
    auto take = [&](size_t bytes) -> void* {
        void* p = (void*)wsp;
        wsp += (bytes + 255) & ~(size_t)255;
        return p;
    };
    __bf16* x_bf16  = (__bf16*)take((size_t)BL * DIM * 2);
    __bf16* WxgT    = (__bf16*)take((size_t)HID * DIM * 2);
    __bf16* WdinT   = (__bf16*)take((size_t)DSSH * DIM * 2);
    __bf16* WgateT  = (__bf16*)take((size_t)HID * DSSH * 2);
    __bf16* WoutT   = (__bf16*)take((size_t)DIM * HID * 2);
    __bf16* CCT     = (__bf16*)take((size_t)DSSH * 1024 * 2);
    __bf16* xg_bf16 = (__bf16*)take((size_t)BL * HID * 2);
    float*  xd      = (float*)take((size_t)BL * DSSH * 4);
    float*  xn      = (float*)take((size_t)BL * DSSH * 4);
    float*  xnT     = (float*)take((size_t)DSSH * BL * 4);
    __bf16* A_s     = (__bf16*)take((size_t)Ll * 1024 * 2);
    float*  KT      = (float*)take((size_t)DSSH * Ll * 4);
    float2* Kf      = (float2*)take((size_t)DSSH * N2 * 8);
    float*  dT      = (float*)take((size_t)DSSH * BL * 4);
    __bf16* d_bf16  = (__bf16*)take((size_t)BL * DSSH * 2);
    __bf16* h_bf16  = (__bf16*)take((size_t)BL * HID * 2);
    float*  lamR    = (float*)take(DSSN * 4);
    float*  lamI    = (float*)take(DSSN * 4);
    float*  wR      = (float*)take(DSSN * 4);
    float*  wI      = (float*)take(DSSN * 4);

    const dim3 tb(32, 8);

    dss_setup<<<2, 256, 0, stream>>>(LR, LI, lamR, lamI, wR, wI);
    cprep_kernel<<<(DSSH * 1024) / 256, 256, 0, stream>>>(CR, CI, CCT);
    transpose_kernel<__bf16><<<dim3(HID / 32, DIM / 32), tb, 0, stream>>>(Wxg, WxgT, DIM, HID);
    transpose_kernel<__bf16><<<dim3(DSSH / 32, DIM / 32), tb, 0, stream>>>(Wdin, WdinT, DIM, DSSH);
    transpose_kernel<__bf16><<<dim3(HID / 32, DSSH / 32), tb, 0, stream>>>(Wgate, WgateT, DSSH, HID);
    transpose_kernel<__bf16><<<dim3(DIM / 32, HID / 32), tb, 0, stream>>>(Wout, WoutT, HID, DIM);

    ln_kernel<<<BL, 256, 0, stream>>>(idx, n1_w, n1_b, nullptr, x_bf16, DIM);

    // xg = x @ Wxg (bf16 out) — TDM-staged when available
#if HAS_TDM
    wmma_gemm_tdm<1><<<dim3(HID / 128, BL / 128), 256, 0, stream>>>(
        x_bf16, WxgT, xg_bf16, nullptr, nullptr, nullptr, BL, HID, DIM);
#else
    wmma_gemm<1><<<dim3(HID / 128, BL / 128), 256, 0, stream>>>(
        x_bf16, WxgT, xg_bf16, nullptr, nullptr, nullptr, BL, HID, DIM);
#endif

    wmma_gemm<0><<<dim3(DSSH / 128, BL / 128), 256, 0, stream>>>(
        x_bf16, WdinT, xd, nullptr, nullptr, nullptr, BL, DSSH, DIM);

    ln_kernel<<<BL, 256, 0, stream>>>(xd, n2_w, n2_b, xn, nullptr, DSSH);
    transpose_kernel<float><<<dim3(DSSH / 32, BL / 32), tb, 0, stream>>>(xn, xnT, BL, DSSH);

    sgen_kernel<<<Ll, 256, 0, stream>>>(lamR, lamI, wR, wI, A_s);
    wmma_gemm<2><<<dim3(DSSH / 128, Ll / 128), 256, 0, stream>>>(
        A_s, CCT, KT, nullptr, nullptr, nullptr, Ll, DSSH, 1024);

    fft_k_kernel<<<DSSH, 256, 0, stream>>>(KT, Kf);
    fft_conv_kernel<<<Bb * DSSH, 256, 0, stream>>>(xnT, Kf, Dp, dT);
    transpose_kernel<__bf16><<<dim3(BL / 32, DSSH / 32), tb, 0, stream>>>(dT, d_bf16, DSSH, BL);

    wmma_gemm<3><<<dim3(HID / 128, BL / 128), 256, 0, stream>>>(
        d_bf16, WgateT, h_bf16, xg_bf16, nullptr, nullptr, BL, HID, DSSH);

    // out = h @ Wout + bout + idx — TDM-staged when available
#if HAS_TDM
    wmma_gemm_tdm<4><<<dim3(DIM / 128, BL / 128), 256, 0, stream>>>(
        h_bf16, WoutT, out, nullptr, bout, idx, BL, DIM, HID);
#else
    wmma_gemm<4><<<dim3(DIM / 128, BL / 128), 256, 0, stream>>>(
        h_bf16, WoutT, out, nullptr, bout, idx, BL, DIM, HID);
#endif
}
